// GINE_Encoder_72086731096840
// MI455X (gfx1250) — matrix operand order repaired and checked by reading, not compiled, not run on hardware
//
#include <hip/hip_runtime.h>
#include <hip/hip_bf16.h>

typedef __attribute__((ext_vector_type(16))) _Float16 v16h;
typedef __attribute__((ext_vector_type(8)))  _Float16 v8h;
typedef __attribute__((ext_vector_type(8)))  float    v8f;

#define DQ    64   // feature dim
#define TPAD  72   // padded f16 row stride (144B, multiple of 16B)
#define APADF 68   // padded f32 row stride (272B, multiple of 16B)

static __device__ __forceinline__ v16h cat8(v8h lo, v8h hi) {
  return __builtin_shufflevector(lo, hi, 0,1,2,3,4,5,6,7,8,9,10,11,12,13,14,15);
}

// A fragment (16x32 f16) from an f16 LDS tile, row stride TPAD halves.
// ISA 7.12.2: lanes 0-15 hold row m=lane, K {0..7,16..23}; lanes 16-31 K {8..15,24..31}.
static __device__ __forceinline__ v16h load_a_frag(const _Float16* tile, int lane, int kbase) {
  int m    = lane & 15;
  int koff = (lane < 16) ? 0 : 8;
  const _Float16* row = tile + m * TPAD + kbase + koff;
  v8h lo = *(const v8h*)(row);
  v8h hi = *(const v8h*)(row + 16);
  return cat8(lo, hi);
}

// A fragment built from an f32 LDS tile (async-staged), converting on the fly.
static __device__ __forceinline__ v16h load_a_frag_f32(const float* tile, int lane, int kbase) {
  int m    = lane & 15;
  int koff = (lane < 16) ? 0 : 8;
  const float* row = tile + m * APADF + kbase + koff;
  v16h out;
  #pragma unroll
  for (int i = 0; i < 8; ++i) out[i]     = (_Float16)row[i];
  #pragma unroll
  for (int i = 0; i < 8; ++i) out[8 + i] = (_Float16)row[16 + i];
  return out;
}

// B fragment (32x16 f16) from LDS weight stored transposed Bt[n][k], row stride DQ halves.
static __device__ __forceinline__ v16h load_b_frag(const _Float16* bt, int lane, int kbase, int nbase) {
  int n    = nbase + (lane & 15);
  int kofs = (lane < 16) ? 0 : 8;
  const _Float16* row = bt + n * DQ + kbase + kofs;
  v8h lo = *(const v8h*)(row);
  v8h hi = *(const v8h*)(row + 16);
  return cat8(lo, hi);
}

// ---------------- Kernel 1: fused edge GEMM + gather + ReLU + scatter-add ----------------
__global__ __launch_bounds__(256)
void gine_edge_kernel(const float* __restrict__ h,
                      const float* __restrict__ edge_attr,
                      const float* __restrict__ We,   // [64][64] row-major [k][n]
                      const float* __restrict__ be,   // [64]
                      const int*   __restrict__ src,
                      const int*   __restrict__ dst,
                      float*       __restrict__ agg,
                      int E)
{
  __shared__ alignas(16) _Float16 sBt[DQ * DQ];        // We transposed [n][k]  (8 KB)
  __shared__ alignas(16) float    sAf[8][16 * APADF];  // per-wave f32 edge tiles (34 KB)
  __shared__ int sSrc[8][16];
  __shared__ int sDst[8][16];

  const int tid  = threadIdx.x;
  const int lane = tid & 31;
  const int w    = tid >> 5;

  // Stage weight (f32 -> f16, transposed) once per block.
  for (int t = tid; t < DQ * DQ; t += 256) {
    int k = t >> 6, n = t & 63;
    sBt[n * DQ + k] = (_Float16)We[t];
  }
  __syncthreads();

  const int tile = blockIdx.x * 8 + w;
  const int e0   = tile * 16;
  if (e0 + 16 <= E) {
    if (lane < 16) {
      sSrc[w][lane] = src[e0 + lane];
      sDst[w][lane] = dst[e0 + lane];
    }

    // Async-stage the contiguous 16x64 f32 tile (4 KB) into per-wave LDS,
    // bank-padded rows (APADF). 8 x b128 per wave, tracked by ASYNCcnt.
    {
      const unsigned ldsbase = (unsigned)(uintptr_t)(&sAf[w][0]);
      const unsigned lrow = (unsigned)(lane >> 4);        // 0 or 1 (row within 512B chunk)
      const unsigned lcol = (unsigned)((lane & 15) * 16); // byte column within row
      const unsigned long long gbase =
          (unsigned long long)(uintptr_t)(edge_attr + (size_t)e0 * DQ) +
          (unsigned long long)(lane * 16);
      #pragma unroll
      for (int j = 0; j < 8; ++j) {
        unsigned laddr = ldsbase + (unsigned)(2 * j + lrow) * (APADF * 4) + lcol;
        unsigned long long ga = gbase + (unsigned long long)(j * 512);
        asm volatile("global_load_async_to_lds_b128 %0, %1, off"
                     :: "v"(laddr), "v"(ga) : "memory");
      }
      asm volatile("s_wait_asynccnt 0x0" ::: "memory");
    }
    __builtin_amdgcn_wave_barrier();  // LDS ops in-order within a wave

    const int n_lo  = lane & 15;
    const int m_off = (lane < 16) ? 0 : 8;

    for (int nt = 0; nt < 4; ++nt) {
      const int nbase = nt * 16;
      const float bias = be[nbase + n_lo];
      v8f acc;
      #pragma unroll
      for (int r = 0; r < 8; ++r) acc[r] = bias;
      #pragma unroll
      for (int kt = 0; kt < 2; ++kt) {
        v16h a = load_a_frag_f32(sAf[w], lane, kt * 32);
        v16h b = load_b_frag(sBt, lane, kt * 32, nbase);
        acc = __builtin_amdgcn_wmma_f32_16x16x32_f16(false, a, false, b,
                                                     (short)0, acc, false, false);
      }
      // message m = relu(h[src] + e); sum-aggregate at dst (coalesced per row)
      #pragma unroll
      for (int r = 0; r < 8; ++r) {
        const int m  = m_off + r;
        const int sm = sSrc[w][m];
        const int dm = sDst[w][m];
        float v = acc[r] + h[(size_t)sm * DQ + nbase + n_lo];
        v = v > 0.f ? v : 0.f;
        atomicAdd(&agg[(size_t)dm * DQ + nbase + n_lo], v);
      }
    }
  }
}

// ---------------- Kernel 2: z=h+agg -> relu(z@W1+b1)@W2+b2 -> relu -> zr + BN stats ----------------
__global__ __launch_bounds__(256)
void gine_node_mlp_kernel(const float* __restrict__ h,
                          const float* __restrict__ agg,
                          const float* __restrict__ W1, const float* __restrict__ b1,
                          const float* __restrict__ W2, const float* __restrict__ b2,
                          float* __restrict__ zr,
                          float* __restrict__ gsum, float* __restrict__ gsq,
                          int N)
{
  __shared__ alignas(16) _Float16 sW1t[DQ * DQ];
  __shared__ alignas(16) _Float16 sW2t[DQ * DQ];
  __shared__ alignas(16) _Float16 sZ[8][16 * TPAD];
  __shared__ alignas(16) _Float16 sT[8][16 * TPAD];
  __shared__ float lsum[DQ];
  __shared__ float lsq[DQ];

  const int tid  = threadIdx.x;
  const int lane = tid & 31;
  const int w    = tid >> 5;

  for (int t = tid; t < DQ * DQ; t += 256) {
    int k = t >> 6, n = t & 63;
    sW1t[n * DQ + k] = (_Float16)W1[t];
    sW2t[n * DQ + k] = (_Float16)W2[t];
  }
  if (tid < DQ) { lsum[tid] = 0.f; lsq[tid] = 0.f; }
  __syncthreads();

  const int tile = blockIdx.x * 8 + w;
  const int nb   = tile * 16;
  const int n_lo  = lane & 15;
  const int m_off = (lane < 16) ? 0 : 8;

  if (nb + 16 <= N) {
    for (int t = lane; t < 16 * DQ; t += 32) {
      int r = t >> 6, c = t & 63;
      size_t gi = (size_t)(nb + r) * DQ + c;
      sZ[w][r * TPAD + c] = (_Float16)(h[gi] + agg[gi]);
    }
    __builtin_amdgcn_wave_barrier();

    // GEMM1: relu(z @ W1 + b1) -> sT
    for (int nt = 0; nt < 4; ++nt) {
      const int nbase = nt * 16;
      const float bias = b1[nbase + n_lo];
      v8f acc;
      #pragma unroll
      for (int r = 0; r < 8; ++r) acc[r] = bias;
      #pragma unroll
      for (int kt = 0; kt < 2; ++kt) {
        v16h a = load_a_frag(sZ[w], lane, kt * 32);
        v16h b = load_b_frag(sW1t,  lane, kt * 32, nbase);
        acc = __builtin_amdgcn_wmma_f32_16x16x32_f16(false, a, false, b,
                                                     (short)0, acc, false, false);
      }
      #pragma unroll
      for (int r = 0; r < 8; ++r) {
        float v = acc[r] > 0.f ? acc[r] : 0.f;
        sT[w][(m_off + r) * TPAD + nbase + n_lo] = (_Float16)v;
      }
    }
    __builtin_amdgcn_wave_barrier();

    // GEMM2: relu(t @ W2 + b2) -> zr, accumulate BatchNorm sums
    for (int nt = 0; nt < 4; ++nt) {
      const int nbase = nt * 16;
      const float bias = b2[nbase + n_lo];
      v8f acc;
      #pragma unroll
      for (int r = 0; r < 8; ++r) acc[r] = bias;
      #pragma unroll
      for (int kt = 0; kt < 2; ++kt) {
        v16h a = load_a_frag(sT[w], lane, kt * 32);
        v16h b = load_b_frag(sW2t,  lane, kt * 32, nbase);
        acc = __builtin_amdgcn_wmma_f32_16x16x32_f16(false, a, false, b,
                                                     (short)0, acc, false, false);
      }
      float s = 0.f, s2 = 0.f;
      #pragma unroll
      for (int r = 0; r < 8; ++r) {
        float v = acc[r] > 0.f ? acc[r] : 0.f;
        zr[(size_t)(nb + m_off + r) * DQ + nbase + n_lo] = v;
        s += v; s2 += v * v;
      }
      atomicAdd(&lsum[nbase + n_lo], s);
      atomicAdd(&lsq [nbase + n_lo], s2);
    }
  }
  __syncthreads();
  if (tid < DQ) {
    atomicAdd(&gsum[tid], lsum[tid]);
    atomicAdd(&gsq [tid], lsq [tid]);
  }
}

// ---------------- Kernel 3: BatchNorm (batch stats) + h out + global_add_pool ----------------
__global__ __launch_bounds__(256)
void gine_bn_pool_kernel(const float* __restrict__ zr,
                         const float* __restrict__ gsum, const float* __restrict__ gsq,
                         const float* __restrict__ gamma, const float* __restrict__ beta,
                         const int*   __restrict__ batch,
                         float* __restrict__ hout,
                         float* __restrict__ pooled,   // already offset by l*64 columns
                         int pstride, int N)
{
  const int idx = blockIdx.x * 256 + threadIdx.x;
  if (idx >= N * DQ) return;
  const int i = idx >> 6;
  const int c = idx & 63;
  const float invN = 1.0f / (float)N;
  const float mu   = gsum[c] * invN;
  const float var  = gsq[c] * invN - mu * mu;
  const float rstd = rsqrtf(var + 1e-5f);
  const float v = gamma[c] * ((zr[idx] - mu) * rstd) + beta[c];
  hout[idx] = v;
  atomicAdd(&pooled[(size_t)batch[i] * pstride + c], v);
}

// ---------------- Host launcher ----------------
extern "C" void kernel_launch(void* const* d_in, const int* in_sizes, int n_in,
                              void* d_out, int out_size, void* d_ws, size_t ws_size,
                              hipStream_t stream) {
  const float* x          = (const float*)d_in[0];
  const float* edge_attr  = (const float*)d_in[1];
  const float* We         = (const float*)d_in[2];
  const float* be         = (const float*)d_in[3];
  const float* W1         = (const float*)d_in[4];
  const float* b1         = (const float*)d_in[5];
  const float* W2         = (const float*)d_in[6];
  const float* b2         = (const float*)d_in[7];
  const float* gamma      = (const float*)d_in[8];
  const float* beta       = (const float*)d_in[9];
  const int*   edge_index = (const int*)d_in[10];
  const int*   batch      = (const int*)d_in[11];

  const int N = in_sizes[0] / DQ;
  const int E = in_sizes[1] / DQ;
  const int L = in_sizes[2] / (DQ * DQ);
  const int G = (out_size - N * DQ) / (L * DQ);

  float* h    = (float*)d_ws;
  float* agg  = h    + (size_t)N * DQ;
  float* zrb  = agg  + (size_t)N * DQ;
  float* gsum = zrb  + (size_t)N * DQ;
  float* gsq  = gsum + DQ;

  float* out = (float*)d_out;
  hipMemsetAsync(out, 0, (size_t)G * L * DQ * sizeof(float), stream);

  const int* srcp = edge_index;
  const int* dstp = edge_index + E;

  const int eblocks = ((E / 16) + 7) / 8;
  const int nblocks = ((N / 16) + 7) / 8;
  const int pblocks = (N * DQ + 255) / 256;

  const float* hcur = x;
  for (int l = 0; l < L; ++l) {
    hipMemsetAsync(agg,  0, (size_t)N * DQ * sizeof(float), stream);
    hipMemsetAsync(gsum, 0, 2 * DQ * sizeof(float), stream);

    gine_edge_kernel<<<eblocks, 256, 0, stream>>>(
        hcur, edge_attr, We + (size_t)l * DQ * DQ, be + l * DQ, srcp, dstp, agg, E);

    gine_node_mlp_kernel<<<nblocks, 256, 0, stream>>>(
        hcur, agg, W1 + (size_t)l * DQ * DQ, b1 + l * DQ,
        W2 + (size_t)l * DQ * DQ, b2 + l * DQ, zrb, gsum, gsq, N);

    float* hnext = (l == L - 1) ? (out + (size_t)G * L * DQ) : h;
    gine_bn_pool_kernel<<<pblocks, 256, 0, stream>>>(
        zrb, gsum, gsq, gamma + l * DQ, beta + l * DQ, batch,
        hnext, out + l * DQ, L * DQ, N);
    hcur = hnext;
  }
}
